// MultiHeaAttention_43559558316801
// MI455X (gfx1250) — matrix operand order repaired and checked by reading, not compiled, not run on hardware
//
#include <hip/hip_runtime.h>
#include <hip/hip_bf16.h>
#include <cstdint>
#include <cstddef>

// ---- problem constants (B=2, L=2048, D=1024, H=16, DK=DV=64) ----
#define BB 2
#define LL 2048
#define DD 1024
#define HH 16
#define DKV 64
#define CHK 32          // keys per staged chunk
#define NCH (LL / CHK)  // 64 chunks

typedef __attribute__((ext_vector_type(16))) __bf16 v16bf;
typedef __attribute__((ext_vector_type(8)))  __bf16 v8bf;
typedef __attribute__((ext_vector_type(8)))  float  v8f;

// round-to-nearest-even f32 -> bf16 bits
__device__ __forceinline__ unsigned short f2bfu(float f) {
  union { float f; unsigned u; } c; c.f = f;
  unsigned u = c.u + 0x7FFFu + ((c.u >> 16) & 1u);
  return (unsigned short)(u >> 16);
}

// A-matrix 16x32 bf16 fragment (ISA 7.12.2): lane m (0-15): K 0-7 & 16-23,
// lane m+16: K 8-15 & 24-31.  p = row base + k0 (element units).
__device__ __forceinline__ v16bf load_afrag(const unsigned short* p, int ah) {
  v8bf lo = *reinterpret_cast<const v8bf*>(p + ah * 8);
  v8bf hi = *reinterpret_cast<const v8bf*>(p + 16 + ah * 8);
  return __builtin_shufflevector(lo, hi, 0, 1, 2, 3, 4, 5, 6, 7,
                                         8, 9, 10, 11, 12, 13, 14, 15);
}

// B-matrix 32x16 bf16 fragment: lane n holds column n; lanes 0-15: K 0-7 & 16-23,
// lanes 16-31: K 8-15 & 24-31.  p = column base (element units).
__device__ __forceinline__ v16bf load_bfrag(const unsigned short* p, int ah) {
  v8bf lo = *reinterpret_cast<const v8bf*>(p + ah * 8);
  v8bf hi = *reinterpret_cast<const v8bf*>(p + 16 + ah * 8);
  return __builtin_shufflevector(lo, hi, 0, 1, 2, 3, 4, 5, 6, 7,
                                         8, 9, 10, 11, 12, 13, 14, 15);
}

__device__ __forceinline__ v8f wmma_bf16(v16bf a, v16bf b, v8f c) {
  return __builtin_amdgcn_wmma_f32_16x16x32_bf16(
      /*neg_a=*/false, a, /*neg_b=*/false, b,
      /*c_mod=*/(short)0, c, /*reuse_a=*/false, /*reuse_b=*/false);
}

// ---- gfx1250 async copy: global -> LDS, 16B per lane, tracked by ASYNCcnt ----
__device__ __forceinline__ void async_ld_b128(unsigned lds_off,
                                              const unsigned short* gsrc) {
  asm volatile("global_load_async_to_lds_b128 %0, %1, off"
               :: "v"(lds_off), "v"((unsigned long long)(uintptr_t)gsrc)
               : "memory");
}
__device__ __forceinline__ void wait_async0() {
  asm volatile("s_wait_asynccnt 0x0" ::: "memory");
}

// stage a contiguous K chunk: 32 keys x 64 feats bf16 = 4096B (128 thr x 32B)
__device__ __forceinline__ void stage_k_chunk(unsigned ldsbase,
                                              const unsigned short* kp, int c,
                                              int tid) {
  const unsigned short* src = kp + (size_t)c * (CHK * DKV) + tid * 16;
  async_ld_b128(ldsbase + tid * 32, src);
  async_ld_b128(ldsbase + tid * 32 + 16, src + 8);
}
// stage a V^T chunk: 64 feat-rows x 32 keys bf16 (row stride L in global)
__device__ __forceinline__ void stage_v_chunk(unsigned ldsbase,
                                              const unsigned short* vp, int c,
                                              int tid) {
  const int d = tid >> 1, hf = tid & 1;
  const unsigned short* src = vp + (size_t)d * LL + c * CHK + hf * 16;
  const unsigned dst = ldsbase + (unsigned)(d * CHK + hf * 16) * 2;
  async_ld_b128(dst, src);
  async_ld_b128(dst + 16, src + 8);
}

// score tile (16q x 16k) from LDS-resident K chunk
__device__ __forceinline__ v8f score_tile(const unsigned short* kb, int sub,
                                          int am, int ah, v16bf qa0, v16bf qa1) {
  const unsigned short* kr = kb + (sub * 16 + am) * DKV;
  v8f s = {};
  s = wmma_bf16(qa0, load_bfrag(kr, ah), s);
  s = wmma_bf16(qa1, load_bfrag(kr + 32, ah), s);
  return s;
}

// ---------------- f32 -> bf16 bulk convert ----------------
__global__ void f32_to_bf16_kernel(const float* __restrict__ src,
                                   unsigned short* __restrict__ dst, int n) {
  for (int i = blockIdx.x * blockDim.x + threadIdx.x; i < n;
       i += gridDim.x * blockDim.x)
    dst[i] = f2bfu(src[i]);
}

// ---------------- GEMM: Y = X[4096,1024] @ W[1024,1024]^T ----------------
// mode 0: bf16 out, head-split [B,H,L,64]; mode 1: bf16 out [B,H,64,L];
// mode 2: f32 out row-major [4096,1024]
__global__ __launch_bounds__(32) void gemm_xwT_bf16(
    const unsigned short* __restrict__ X, const unsigned short* __restrict__ W,
    unsigned short* __restrict__ obf, float* __restrict__ of32, int mode) {
  const int lane = threadIdx.x & 31;
  const int am = lane & 15, ah = lane >> 4;
  const int mt = blockIdx.x;   // 0..255
  const int nt = blockIdx.y;   // 0..15
  const unsigned short* xrow = X + (size_t)(mt * 16 + am) * DD;

  v8f acc[4] = {};
#pragma unroll 4
  for (int k0 = 0; k0 < DD; k0 += 32) {
    v16bf a = load_afrag(xrow + k0, ah);
#pragma unroll
    for (int j = 0; j < 4; ++j) {
      v16bf b = load_bfrag(W + (size_t)(nt * 64 + j * 16 + am) * DD + k0, ah);
      acc[j] = wmma_bf16(a, b, acc[j]);
    }
  }

#pragma unroll
  for (int j = 0; j < 4; ++j)
#pragma unroll
    for (int r = 0; r < 8; ++r) {
      const int row = r + ah * 8;
      const int m = mt * 16 + row;
      const int col = nt * 64 + j * 16 + am;
      const float v = acc[j][r];
      if (mode == 2) {
        of32[(size_t)m * DD + col] = v;
      } else {
        const int b = m >> 11, l = m & (LL - 1);
        const int h = col >> 6, d = col & (DKV - 1);
        const size_t idx =
            (mode == 0) ? ((size_t)((b * HH + h) * LL + l) * DKV + d)
                        : ((size_t)((b * HH + h) * DKV + d) * LL + l);
        obf[idx] = f2bfu(v);
      }
    }
}

// ---------------- fused attention ----------------
// 4 waves / block: 4 consecutive 16-row q-tiles of one (b,h) share K/V chunks
// staged into double-buffered LDS via global_load_async_to_lds_b128 (4x less
// L2 traffic than per-wave streaming).  3 exact passes (max, sum, emit):
// attn is written to HBM exactly once; pass 3 repacks P through a per-wave
// LDS tile into A-fragments feeding the ctx WMMAs against LDS-resident V^T.
__global__ __launch_bounds__(128) void attn_fused_kernel(
    const unsigned short* __restrict__ qbf,  // [B,H,L,64]
    const unsigned short* __restrict__ kbf,  // [B,H,L,64]
    const unsigned short* __restrict__ vtb,  // [B,H,64,L]
    unsigned short* __restrict__ ctxb,       // [B*L, H*64] bf16
    float* __restrict__ attn)                // [B,H,L,L] f32
{
  __shared__ unsigned short ldsK[2][CHK * DKV];  // 2 x 4KB
  __shared__ unsigned short ldsV[2][DKV * CHK];  // 2 x 4KB
  __shared__ unsigned short psh[4][16 * 32];     // per-wave P repack, 4KB

  const int tid = threadIdx.x;
  const int wv = tid >> 5;
  const int lane = tid & 31;
  const int am = lane & 15, ah = lane >> 4;
  const int bh = blockIdx.x >> 5;              // 32 blocks per (b,h)
  const int qt = (blockIdx.x & 31) * 4 + wv;   // q-tile of this wave

  const unsigned short* qp = qbf + ((size_t)bh * LL + qt * 16) * DKV;
  const unsigned short* kp = kbf + (size_t)bh * LL * DKV;
  const unsigned short* vp = vtb + (size_t)bh * DKV * LL;
  float* ap = attn + (size_t)bh * LL * LL + (size_t)qt * 16 * LL;
  const float isc = 0.125f;  // 1/sqrt(DK)

  const unsigned kb_[2] = {(unsigned)(uintptr_t)&ldsK[0][0],
                           (unsigned)(uintptr_t)&ldsK[1][0]};
  const unsigned vb_[2] = {(unsigned)(uintptr_t)&ldsV[0][0],
                           (unsigned)(uintptr_t)&ldsV[1][0]};

  // Q block fragments (DK=64 -> two K=32 fragments), resident across passes
  const v16bf qa0 = load_afrag(qp + am * DKV + 0, ah);
  const v16bf qa1 = load_afrag(qp + am * DKV + 32, ah);

  float rm[8], rs[8];
#pragma unroll
  for (int r = 0; r < 8; ++r) { rm[r] = -3.0e38f; rs[r] = 0.0f; }

  // ---- pass 1: exact row max ----
  stage_k_chunk(kb_[0], kp, 0, tid);
  wait_async0();
  __syncthreads();
  for (int c = 0; c < NCH; ++c) {
    const int cur = c & 1;
    if (c + 1 < NCH) stage_k_chunk(kb_[cur ^ 1], kp, c + 1, tid);
    const unsigned short* kbl = (const unsigned short*)ldsK[cur];
#pragma unroll
    for (int sub = 0; sub < 2; ++sub) {
      v8f s = score_tile(kbl, sub, am, ah, qa0, qa1);
#pragma unroll
      for (int r = 0; r < 8; ++r) {
        float v = s[r] * isc;
        v = fmaxf(v, __shfl_xor(v, 1));
        v = fmaxf(v, __shfl_xor(v, 2));
        v = fmaxf(v, __shfl_xor(v, 4));
        v = fmaxf(v, __shfl_xor(v, 8));  // stays inside each 16-lane half
        rm[r] = fmaxf(rm[r], v);
      }
    }
    wait_async0();
    __syncthreads();
  }

  // ---- pass 2: row sum of exp ----
  stage_k_chunk(kb_[0], kp, 0, tid);
  wait_async0();
  __syncthreads();
  for (int c = 0; c < NCH; ++c) {
    const int cur = c & 1;
    if (c + 1 < NCH) stage_k_chunk(kb_[cur ^ 1], kp, c + 1, tid);
    const unsigned short* kbl = (const unsigned short*)ldsK[cur];
#pragma unroll
    for (int sub = 0; sub < 2; ++sub) {
      v8f s = score_tile(kbl, sub, am, ah, qa0, qa1);
#pragma unroll
      for (int r = 0; r < 8; ++r) {
        float e = __expf(s[r] * isc - rm[r]);
        e += __shfl_xor(e, 1);
        e += __shfl_xor(e, 2);
        e += __shfl_xor(e, 4);
        e += __shfl_xor(e, 8);
        rs[r] += e;
      }
    }
    wait_async0();
    __syncthreads();
  }
  float ri[8];
#pragma unroll
  for (int r = 0; r < 8; ++r) ri[r] = 1.0f / rs[r];

  // ---- pass 3: store normalized attn once + accumulate ctx = P @ V ----
  v8f cacc[4] = {};
  stage_k_chunk(kb_[0], kp, 0, tid);
  stage_v_chunk(vb_[0], vp, 0, tid);
  wait_async0();
  __syncthreads();
  for (int c = 0; c < NCH; ++c) {
    const int cur = c & 1;
    if (c + 1 < NCH) {
      stage_k_chunk(kb_[cur ^ 1], kp, c + 1, tid);
      stage_v_chunk(vb_[cur ^ 1], vp, c + 1, tid);
    }
    const unsigned short* kbl = (const unsigned short*)ldsK[cur];
    const unsigned short* vbl = (const unsigned short*)ldsV[cur];
    v8f s0 = score_tile(kbl, 0, am, ah, qa0, qa1);
    v8f s1 = score_tile(kbl, 1, am, ah, qa0, qa1);
#pragma unroll
    for (int r = 0; r < 8; ++r) {
      const int row = r + ah * 8;
      const float p0 = __expf(s0[r] * isc - rm[r]) * ri[r];
      const float p1 = __expf(s1[r] * isc - rm[r]) * ri[r];
      ap[(size_t)row * LL + c * CHK + am] = p0;
      ap[(size_t)row * LL + c * CHK + 16 + am] = p1;
      psh[wv][row * 32 + am] = f2bfu(p0);        // C-layout -> A-layout repack
      psh[wv][row * 32 + 16 + am] = f2bfu(p1);
    }
    asm volatile("s_wait_dscnt 0" ::: "memory");  // per-wave LDS RAW handoff
    const v16bf pa = load_afrag((const unsigned short*)psh[wv] + am * 32, ah);
#pragma unroll
    for (int j = 0; j < 4; ++j) {
      v16bf bv = load_bfrag(vbl + (j * 16 + am) * CHK, ah);
      cacc[j] = wmma_bf16(pa, bv, cacc[j]);
    }
    wait_async0();
    __syncthreads();
  }

  // ---- store ctx tile (merged-head layout for final GEMM) ----
  const int b = bh / HH, h = bh % HH;
#pragma unroll
  for (int j = 0; j < 4; ++j)
#pragma unroll
    for (int r = 0; r < 8; ++r) {
      const int row = r + ah * 8;
      const size_t m = (size_t)b * LL + qt * 16 + row;
      ctxb[m * DD + h * DKV + j * 16 + am] = f2bfu(cacc[j][r]);
    }
}

// ---------------- host launcher ----------------
extern "C" void kernel_launch(void* const* d_in, const int* in_sizes, int n_in,
                              void* d_out, int out_size, void* d_ws,
                              size_t ws_size, hipStream_t stream) {
  (void)in_sizes; (void)n_in; (void)out_size; (void)ws_size;
  // setup_inputs order: key, value, query, wq, wk, wv, wo
  const float* key = (const float*)d_in[0];
  const float* value = (const float*)d_in[1];
  const float* query = (const float*)d_in[2];
  const float* wq = (const float*)d_in[3];
  const float* wk = (const float*)d_in[4];
  const float* wv = (const float*)d_in[5];
  const float* wo = (const float*)d_in[6];

  const size_t XN = (size_t)BB * LL * DD;  // 4,194,304
  const size_t WN = (size_t)DD * DD;       // 1,048,576

  // workspace: bf16 copies (inputs, weights) + bf16 q/k/vT/ctx  (~64 MB)
  unsigned short* ws = (unsigned short*)d_ws;
  unsigned short* xq = ws;
  unsigned short* xk = xq + XN;
  unsigned short* xv = xk + XN;
  unsigned short* wqb = xv + XN;
  unsigned short* wkb = wqb + WN;
  unsigned short* wvb = wkb + WN;
  unsigned short* wob = wvb + WN;
  unsigned short* qb = wob + WN;   // [B,H,L,64]
  unsigned short* kb = qb + XN;    // [B,H,L,64]
  unsigned short* vt = kb + XN;    // [B,H,64,L]
  unsigned short* cx = vt + XN;    // [4096,1024]

  float* out_f = (float*)d_out;    // out: 4,194,304 f32
  float* attn = out_f + XN;        // attn: 134,217,728 f32

  const dim3 cblk(256), cgrid(2048);
  hipLaunchKernelGGL(f32_to_bf16_kernel, cgrid, cblk, 0, stream, query, xq, (int)XN);
  hipLaunchKernelGGL(f32_to_bf16_kernel, cgrid, cblk, 0, stream, key,   xk, (int)XN);
  hipLaunchKernelGGL(f32_to_bf16_kernel, cgrid, cblk, 0, stream, value, xv, (int)XN);
  hipLaunchKernelGGL(f32_to_bf16_kernel, cgrid, cblk, 0, stream, wq, wqb, (int)WN);
  hipLaunchKernelGGL(f32_to_bf16_kernel, cgrid, cblk, 0, stream, wk, wkb, (int)WN);
  hipLaunchKernelGGL(f32_to_bf16_kernel, cgrid, cblk, 0, stream, wv, wvb, (int)WN);
  hipLaunchKernelGGL(f32_to_bf16_kernel, cgrid, cblk, 0, stream, wo, wob, (int)WN);

  const dim3 ggrid(BB * LL / 16, DD / 64), gblk(32);  // (256,16) x 1 wave
  hipLaunchKernelGGL(gemm_xwT_bf16, ggrid, gblk, 0, stream, xq, wqb, qb, (float*)nullptr, 0);
  hipLaunchKernelGGL(gemm_xwT_bf16, ggrid, gblk, 0, stream, xk, wkb, kb, (float*)nullptr, 0);
  hipLaunchKernelGGL(gemm_xwT_bf16, ggrid, gblk, 0, stream, xv, wvb, vt, (float*)nullptr, 1);

  // 4 q-tiles per block -> B*H*(L/16)/4 = 1024 blocks of 128 threads
  hipLaunchKernelGGL(attn_fused_kernel, dim3(BB * HH * (LL / 16) / 4), dim3(128),
                     0, stream, qb, kb, vt, cx, attn);

  hipLaunchKernelGGL(gemm_xwT_bf16, ggrid, gblk, 0, stream, cx, wob,
                     (unsigned short*)nullptr, out_f, 2);
}
